// LSTM_20598663151661
// MI455X (gfx1250) — compile-verified, matrix operand order checked by reading, not run on hardware
//
#include <hip/hip_runtime.h>
#include <hip/hip_bf16.h>
#include <math.h>

typedef __attribute__((ext_vector_type(16))) _Float16 v16h;
typedef __attribute__((ext_vector_type(8)))  float    v8f;

#define HID   512
#define DIN   10
#define NCLS  10
#define BATCH 256
#define TSTEP 256
#define KPAD  544           // 512 (h) + 10 (x) + 1 (bias) padded to 17*32
#define NKT   17            // KPAD / 32
#define NMT   128           // 4*HID / 16 rows of gate outputs
#define BT    16            // batch columns per workgroup
#define FRAG_HALFS 512      // 32 lanes * 16 halves per 16x32 fragment k-tile

// ---------------------------------------------------------------------------
// Pack f32 weights -> f16, pre-swizzled into per-lane WMMA A-fragment order.
// A 16x32 f16 fragment (ISA 7.12.2): lane L holds row M = L%16; half h maps to
//   K = ((L<16)?0:8) + (h/8)*16 + (h%8)
// Fragment index = m_tile*NKT + k_tile ; each fragment = 32 lanes * 16 halves.
// Row blocks: [0,512)=g  [512,1024)=i  [1024,1536)=f  [1536,2048)=o
// K blocks:   [0,512)=W_h  [512,522)=W_x  522=bias  [523,544)=0
// ---------------------------------------------------------------------------
__global__ void lstm_pack_w(const float* wgx, const float* wgh, const float* bg,
                            const float* wix, const float* wih, const float* bi,
                            const float* wfx, const float* wfh, const float* bf,
                            const float* wox, const float* woh, const float* bo,
                            _Float16* __restrict__ wpack) {
  int tid  = blockIdx.x * blockDim.x + threadIdx.x;
  int frag = tid >> 5;
  int lane = tid & 31;
  if (frag >= NMT * NKT) return;
  int m_tile = frag / NKT;
  int k_tile = frag % NKT;
  int gate   = m_tile >> 5;                        // 0..3
  int r      = ((m_tile & 31) << 4) + (lane & 15); // row within gate 0..511

  const float* wh = (gate == 0) ? wgh : (gate == 1) ? wih : (gate == 2) ? wfh : woh;
  const float* wx = (gate == 0) ? wgx : (gate == 1) ? wix : (gate == 2) ? wfx : wox;
  const float* bb = (gate == 0) ? bg  : (gate == 1) ? bi  : (gate == 2) ? bf  : bo;

  int kbase = (lane < 16) ? 0 : 8;
  v16h out;
  #pragma unroll
  for (int h = 0; h < 16; ++h) {
    int K = k_tile * 32 + kbase + (h >> 3) * 16 + (h & 7);
    float v = 0.0f;
    if (K < HID)              v = wh[r * HID + K];
    else if (K < HID + DIN)   v = wx[r * DIN + (K - HID)];
    else if (K == HID + DIN)  v = bb[r];
    out[h] = (_Float16)v;
  }
  *(v16h*)(wpack + (size_t)(frag * 32 + lane) * 16) = out;
}

// Branchless activations: keep the gate math on v_exp_f32/v_rcp_f32 so it
// co-executes with the (TRANS-tracked) 16-bit WMMA pipe instead of doing
// libm's EXEC-mask range splits.
__device__ __forceinline__ float sigf(float x) {
  return __builtin_amdgcn_rcpf(1.0f + __expf(-x));
}
__device__ __forceinline__ float tanh_fast(float x) {
  return 2.0f * sigf(2.0f * x) - 1.0f;
}

// B-fragment inverse mapping: global K row k (0..543), column n (0..15) lives
// at lane = n + ((k%16)>=8 ? 16:0), half = ((k%32)>=16 ? 8:0) + (k%8),
// inside k-tile block (k/32).
__device__ __forceinline__ int bfrag_off(int k, int n) {
  int kt = k >> 5, k32 = k & 31;
  int lw = n + (((k32 & 15) >= 8) ? 16 : 0);
  int hw = ((k32 >= 16) ? 8 : 0) + (k32 & 7);
  return kt * FRAG_HALFS + lw * 16 + hw;
}

#define WMMA_F16(d, a, b) \
  d = __builtin_amdgcn_wmma_f32_16x16x32_f16(false, (a), false, (b), (short)0, (d), false, false)

// ---------------------------------------------------------------------------
// Persistent batch-tile LSTM: 16 workgroups x 16 batch columns, no grid sync.
// 8 waves; wave w owns gate rows [w*64, w*64+64) across all 4 gates.
// Per step: 17 k-tiles x 4 gates WMMA f32_16x16x32_f16; h double-buffered in
// LDS in B-fragment order; c held in registers. Weights are re-streamed from
// L2 every step. An opaque *integer offset* defeats cross-step LICM of the
// weight loads while preserving global address-space provenance, so they
// lower to global_load_b128 (not flat_load) and never spill.
// ---------------------------------------------------------------------------
__global__ __launch_bounds__(256)
void lstm_wmma(const float* __restrict__ x, const _Float16* __restrict__ wpack,
               const float* __restrict__ wout, const float* __restrict__ bout,
               float* __restrict__ out) {
  __shared__ _Float16 hbuf[2][16 * FRAG_HALFS];  // k-tiles 0..15 (h rows), x2 buffers
  __shared__ _Float16 xblk[FRAG_HALFS];          // k-tile 16 (x rows + bias + zero pad)
  __shared__ float    pbuf[BT * 12];

  const int tid   = threadIdx.x;
  const int lane  = tid & 31;
  const int wave  = tid >> 5;
  const int bcol0 = blockIdx.x * BT;

  for (int i = tid; i < 16 * FRAG_HALFS; i += 256) hbuf[0][i] = (_Float16)0.0f;
  for (int i = tid; i < FRAG_HALFS;      i += 256) xblk[i]    = (_Float16)0.0f;
  __syncthreads();
  if (tid < BT) {                                   // bias row: K=522 -> k32=10
    xblk[bfrag_off(HID + DIN, tid) - 16 * FRAG_HALFS] = (_Float16)1.0f;
  }

  v8f creg[4];
  #pragma unroll
  for (int m = 0; m < 4; ++m)
    #pragma unroll
    for (int i = 0; i < 8; ++i) creg[m][i] = 0.0f;

  int cur = 0;
  for (int t = 0; t < TSTEP; ++t) {
    // stage x_t into the augmented k-tile (rows 512..521)
    if (tid < DIN * BT) {
      int d = tid / BT, col = tid % BT;
      float v = x[((size_t)(bcol0 + col) * TSTEP + t) * DIN + d];
      int ln = (d >= 8) ? (16 + col) : col;        // bfrag mapping for k32 = d
      xblk[ln * 16 + (d & 7)] = (_Float16)v;
    }
    __syncthreads();
    const int nxt = cur ^ 1;

    // Opaque zero offset: loads can't be proven loop-invariant across t,
    // but wpack's addrspace(1) provenance survives -> global_load_b128.
    size_t woff = 0;
    asm volatile("" : "+s"(woff));
    const _Float16* wp = wpack + woff;

    #pragma unroll
    for (int mt = 0; mt < 4; ++mt) {
      const int ht = wave * 4 + mt;                // h-row tile 0..31
      v8f accg = {}, acci = {}, accf = {}, acco = {};

      const _Float16* pg = wp + (size_t)((0 * 32 + ht) * NKT) * FRAG_HALFS + lane * 16;
      const _Float16* pi = wp + (size_t)((1 * 32 + ht) * NKT) * FRAG_HALFS + lane * 16;
      const _Float16* pf = wp + (size_t)((2 * 32 + ht) * NKT) * FRAG_HALFS + lane * 16;
      const _Float16* po = wp + (size_t)((3 * 32 + ht) * NKT) * FRAG_HALFS + lane * 16;
      const _Float16* hb = &hbuf[cur][lane * 16];

      #pragma unroll 2
      for (int kt = 0; kt < 16; ++kt) {
        v16h bfrag = *(const v16h*)(hb + kt * FRAG_HALFS);
        v16h ag = *(const v16h*)(pg + kt * FRAG_HALFS);
        v16h ai = *(const v16h*)(pi + kt * FRAG_HALFS);
        v16h af = *(const v16h*)(pf + kt * FRAG_HALFS);
        v16h ao = *(const v16h*)(po + kt * FRAG_HALFS);
        __builtin_prefetch(pg + (kt + 1) * FRAG_HALFS, 0, 3);  // next k-tile
        WMMA_F16(accg, ag, bfrag);
        WMMA_F16(acci, ai, bfrag);
        WMMA_F16(accf, af, bfrag);
        WMMA_F16(acco, ao, bfrag);
      }
      {                                            // augmented k-tile (x + bias)
        v16h bfrag = *(const v16h*)&xblk[lane * 16];
        v16h ag = *(const v16h*)(pg + 16 * FRAG_HALFS);
        v16h ai = *(const v16h*)(pi + 16 * FRAG_HALFS);
        v16h af = *(const v16h*)(pf + 16 * FRAG_HALFS);
        v16h ao = *(const v16h*)(po + 16 * FRAG_HALFS);
        WMMA_F16(accg, ag, bfrag);
        WMMA_F16(acci, ai, bfrag);
        WMMA_F16(accf, af, bfrag);
        WMMA_F16(acco, ao, bfrag);
      }

      // gates + cell update; write h_new into the other buffer (B-frag order)
      const int col  = lane & 15;
      const int roff = (lane < 16) ? 0 : 8;        // C/D layout: VGPR v -> M=v or v+8
      #pragma unroll
      for (int i = 0; i < 8; ++i) {
        float g  = tanh_fast(accg[i]);
        float ii = sigf(acci[i]);
        float ff = sigf(accf[i]);
        float oo = sigf(acco[i]);
        float cn = g * ii + creg[mt][i] * ff;
        creg[mt][i] = cn;
        float hn = tanh_fast(cn) * oo;
        hbuf[nxt][bfrag_off(ht * 16 + i + roff, col)] = (_Float16)hn;
      }
    }
    __syncthreads();
    cur ^= 1;
  }

  // head: p = w_out @ h + b_out ; softmax over classes
  if (tid < NCLS * BT) {
    int col = tid / NCLS, cls = tid % NCLS;
    float s = bout[cls];
    for (int r = 0; r < HID; ++r)
      s += wout[cls * HID + r] * (float)hbuf[cur][bfrag_off(r, col)];
    pbuf[col * 12 + cls] = s;
  }
  __syncthreads();
  if (tid < BT) {
    int col = tid;
    float m = pbuf[col * 12];
    for (int c = 1; c < NCLS; ++c) m = fmaxf(m, pbuf[col * 12 + c]);
    float e[NCLS], den = 0.0f;
    for (int c = 0; c < NCLS; ++c) { e[c] = __expf(pbuf[col * 12 + c] - m); den += e[c]; }
    float rden = __builtin_amdgcn_rcpf(den);
    for (int c = 0; c < NCLS; ++c)
      out[(size_t)(bcol0 + col) * NCLS + c] = e[c] * rden;
  }
}

extern "C" void kernel_launch(void* const* d_in, const int* in_sizes, int n_in,
                              void* d_out, int out_size, void* d_ws, size_t ws_size,
                              hipStream_t stream) {
  const float* x    = (const float*)d_in[0];
  const float* wgx  = (const float*)d_in[1];
  const float* wgh  = (const float*)d_in[2];
  const float* bg   = (const float*)d_in[3];
  const float* wix  = (const float*)d_in[4];
  const float* wih  = (const float*)d_in[5];
  const float* bi   = (const float*)d_in[6];
  const float* wfx  = (const float*)d_in[7];
  const float* wfh  = (const float*)d_in[8];
  const float* bf   = (const float*)d_in[9];
  const float* wox  = (const float*)d_in[10];
  const float* woh  = (const float*)d_in[11];
  const float* bo   = (const float*)d_in[12];
  const float* wout = (const float*)d_in[13];
  const float* bout = (const float*)d_in[14];

  _Float16* wpack = (_Float16*)d_ws;   // NMT*NKT*32*16 halves = ~2.2 MB

  int packThreads = NMT * NKT * 32;    // 69632
  lstm_pack_w<<<(packThreads + 255) / 256, 256, 0, stream>>>(
      wgx, wgh, bg, wix, wih, bi, wfx, wfh, bf, wox, woh, bo, wpack);

  lstm_wmma<<<BATCH / BT, 256, 0, stream>>>(x, wpack, wout, bout, (float*)d_out);
}
